// AtlasBlock_70385924047329
// MI455X (gfx1250) — matrix operand order repaired and checked by reading, not compiled, not run on hardware
//
#include <hip/hip_runtime.h>
#include <hip/hip_bf16.h>
#include <stdint.h>

typedef _Float16 half_t;
typedef __attribute__((ext_vector_type(16))) _Float16 v16h;
typedef __attribute__((ext_vector_type(8)))  _Float16 v8h;
typedef __attribute__((ext_vector_type(2)))  _Float16 v2h;
typedef __attribute__((ext_vector_type(8)))  float    v8f;
typedef __attribute__((ext_vector_type(4)))  int      v4i;

// ---------------------------------------------------------------- helpers ---
__device__ __forceinline__ v16h cat8(v8h lo, v8h hi) {
  v16h r;
#pragma unroll
  for (int i = 0; i < 8; ++i) { r[i] = lo[i]; r[i + 8] = hi[i]; }
  return r;
}

__device__ __forceinline__ v8f wmma16(v16h a, v16h b, v8f c) {
  // D = A(16x32 f16) * B(32x16 f16) + C(16x16 f32)
  return __builtin_amdgcn_wmma_f32_16x16x32_f16(false, a, false, b,
                                                (short)0, c, false, false);
}

// 16-byte global -> LDS copy; async (ASYNCcnt-tracked) on gfx1250.
__device__ __forceinline__ void async_cp16(const half_t* g, half_t* l) {
#if defined(__AMDGCN__) && __has_builtin(__builtin_amdgcn_global_load_async_to_lds_b128)
  __builtin_amdgcn_global_load_async_to_lds_b128(
      (__attribute__((address_space(1))) v4i*)g,
      (__attribute__((address_space(3))) v4i*)l, 0, 0);
#else
  *(v8h*)l = *(const v8h*)g;
#endif
}

__device__ __forceinline__ void wait_async_lds() {
#if defined(__AMDGCN__) && __has_builtin(__builtin_amdgcn_global_load_async_to_lds_b128)
#if __has_builtin(__builtin_amdgcn_s_wait_asynccnt)
  __builtin_amdgcn_s_wait_asynccnt(0);
#else
  asm volatile("s_wait_asynccnt 0x0" ::: "memory");
#endif
#endif
}

// ------------------------------------------------------------ LayerNorm -----
// one block (256 threads) per row of 256; out is f16
__global__ __launch_bounds__(256) void ln_f16_kernel(
    const float* __restrict__ x, const float* __restrict__ g,
    const float* __restrict__ bb, half_t* __restrict__ out) {
  __shared__ float sh[8];
  const int row = blockIdx.x, t = threadIdx.x;
  const int lane = t & 31, w = t >> 5;
  const float v = x[(size_t)row * 256 + t];

  float s = v;
#pragma unroll
  for (int o = 16; o; o >>= 1) s += __shfl_xor(s, o, 32);
  if (lane == 0) sh[w] = s;
  __syncthreads();
  float tot = (lane < 8) ? sh[lane] : 0.f;
#pragma unroll
  for (int o = 4; o; o >>= 1) tot += __shfl_xor(tot, o, 32);
  tot = __shfl(tot, 0, 32);
  const float mu = tot * (1.f / 256.f);
  const float d = v - mu;
  __syncthreads();

  s = d * d;
#pragma unroll
  for (int o = 16; o; o >>= 1) s += __shfl_xor(s, o, 32);
  if (lane == 0) sh[w] = s;
  __syncthreads();
  tot = (lane < 8) ? sh[lane] : 0.f;
#pragma unroll
  for (int o = 4; o; o >>= 1) tot += __shfl_xor(tot, o, 32);
  tot = __shfl(tot, 0, 32);
  const float var = tot * (1.f / 256.f);

  const float y = d * rsqrtf(var + 1e-5f) * g[t] + bb[t];
  out[(size_t)row * 256 + t] = (half_t)y;
}

// ------------------------------------------------------- generic WMMA GEMM --
// C[M,N] = A[M,K](f16) * W[K,N](f32->f16) + bias ; writes f32 and/or f16 out.
// block = 256 threads (8 waves); block tile 64x64; K step 32; double-buffered
// LDS with async global->LDS staging of the A tile.
__global__ __launch_bounds__(256) void gemm_wmma_kernel(
    const half_t* __restrict__ A, int lda,
    const float* __restrict__ W, int ldw,
    const float* __restrict__ bias,
    float* __restrict__ C32, int ldc32,
    half_t* __restrict__ C16, int ldc16,
    int M, int N, int K) {
  __shared__ __align__(16) half_t As[2][64][32];
  __shared__ __align__(16) half_t Bt[2][64][32];  // transposed W tile: [n][k]

  const int bm0 = blockIdx.y * 64;
  const int bn0 = blockIdx.x * 64;
  const int tid = threadIdx.x;
  const int lane = tid & 31;
  const int wid = tid >> 5;
  const int wm = wid >> 1, wn = wid & 1;
  const int m16 = lane & 15, hl = lane >> 4;

  const int arow = tid & 63, acg = tid >> 6;  // A stage: 64 rows x 4 groups of 8
  const int kp = tid & 15, ng = tid >> 4;     // W stage: 16 k-pairs x 16 groups of 4 cols

  auto stage = [&](int kc, int sb) {
    // A tile: straight f16 copy -> async to LDS
    async_cp16(A + (size_t)(bm0 + arow) * lda + kc + acg * 8,
               &As[sb][arow][acg * 8]);
    // W tile: f32 -> f16 transpose; pack k-pairs so LDS stores are b32
    const float* wp0 = W + (size_t)(kc + 2 * kp) * ldw + bn0 + 4 * ng;
    const float4 r0 = *(const float4*)wp0;
    const float4 r1 = *(const float4*)(wp0 + ldw);
    const float* a0 = &r0.x;
    const float* a1 = &r1.x;
#pragma unroll
    for (int j = 0; j < 4; ++j) {
      v2h p;
      p[0] = (half_t)a0[j];
      p[1] = (half_t)a1[j];
      *(v2h*)&Bt[sb][4 * ng + j][2 * kp] = p;
    }
  };

  v8f c0 = {}, c1 = {};

  stage(0, 0);
  wait_async_lds();
  __syncthreads();

  int buf = 0;
  for (int kc = 0; kc < K; kc += 32) {
    const bool nxt = (kc + 32 < K);
    if (kc + 64 < K) {  // prefetch the tile after next into L2
      __builtin_prefetch(A + (size_t)(bm0 + arow) * lda + (kc + 64) + acg * 8, 0, 3);
      __builtin_prefetch(W + (size_t)(kc + 64 + 2 * kp) * ldw + bn0 + 4 * ng, 0, 3);
    }
    if (nxt) stage(kc + 32, buf ^ 1);  // overlap staging with WMMA below

    // A fragment: lane m=m16, K = e + 16*hl (contiguous)
    const v8h alo = *(const v8h*)&As[buf][wm * 16 + m16][hl * 16];
    const v8h ahi = *(const v8h*)&As[buf][wm * 16 + m16][hl * 16 + 8];
    const v16h a = cat8(alo, ahi);
    // B fragments: lane n=m16, K = e + 16*hl (contiguous in Bt row)
    {
      const half_t* bp = &Bt[buf][wn * 32 + m16][hl * 16];
      const v16h b = cat8(*(const v8h*)bp, *(const v8h*)(bp + 8));
      c0 = wmma16(a, b, c0);
    }
    {
      const half_t* bp = &Bt[buf][wn * 32 + 16 + m16][hl * 16];
      const v16h b = cat8(*(const v8h*)bp, *(const v8h*)(bp + 8));
      c1 = wmma16(a, b, c1);
    }
    wait_async_lds();  // async writes to buf^1 must land before the barrier
    __syncthreads();
    buf ^= 1;
  }

  // epilogue: C layout -> row = base + r + 8*hl, col = lane&15
  const int n0 = bn0 + wn * 32 + m16;
  const int n1 = n0 + 16;
  const float b0v = bias ? bias[n0] : 0.f;
  const float b1v = bias ? bias[n1] : 0.f;
  const int mrow = bm0 + wm * 16 + hl * 8;
#pragma unroll
  for (int r = 0; r < 8; ++r) {
    const float v0 = c0[r] + b0v;
    const float v1 = c1[r] + b1v;
    const size_t ro = (size_t)(mrow + r);
    if (C32) { C32[ro * ldc32 + n0] = v0; C32[ro * ldc32 + n1] = v1; }
    if (C16) { C16[ro * ldc16 + n0] = (half_t)v0; C16[ro * ldc16 + n1] = (half_t)v1; }
  }
}

// ------------------------------------------------ V transpose (for PV WMMA) -
// qkv f16 [2048][768] (v at col 512..767) -> Vt f16 [B][256][1024] (feat-major)
__global__ __launch_bounds__(256) void vtrans_kernel(const half_t* __restrict__ qkv,
                                                     half_t* __restrict__ Vt) {
  const int idx = blockIdx.x * 256 + threadIdx.x;  // 2*256*1024 total
  const int s = idx & 1023;
  const int d = (idx >> 10) & 255;
  const int b = idx >> 18;
  Vt[idx] = qkv[((size_t)(b * 1024 + s)) * 768 + 512 + d];
}

// ------------------------------------- sliding-window flash attention (WMMA) -
// one wave per (b, h, 16-query block). WIN=256, hd=64.
__global__ __launch_bounds__(32) void attn_kernel(const half_t* __restrict__ qkv,
                                                  const half_t* __restrict__ Vt,
                                                  half_t* __restrict__ attn16) {
  __shared__ __align__(16) half_t Pl[16][32];
  const int q0 = blockIdx.x * 16;
  const int h = blockIdx.y, b = blockIdx.z;
  const int lane = threadIdx.x & 31;
  const int m16 = lane & 15, hl = lane >> 4;

  // Q A-fragments (16x64 split into two 16x32 frags)
  v16h aq[2];
  const size_t rowQ = (size_t)(b * 1024 + q0 + m16) * 768 + h * 64;
#pragma unroll
  for (int kc2 = 0; kc2 < 2; ++kc2) {
    const half_t* qp = qkv + rowQ + kc2 * 32;
    aq[kc2] = cat8(*(const v8h*)(qp + hl * 16), *(const v8h*)(qp + hl * 16 + 8));
  }

  v8f o0 = {}, o1 = {}, o2 = {}, o3 = {};
  float rmax[8], rsum[8];
#pragma unroll
  for (int r = 0; r < 8; ++r) { rmax[r] = -1e30f; rsum[r] = 0.f; }

  int lo = q0 - 255; if (lo < 0) lo = 0; lo &= ~31;
  const int hi = q0 + 15;

  for (int kb = lo; kb <= hi; kb += 32) {
    float pbuf[2][8];
    v8f st[2];
#pragma unroll
    for (int sub = 0; sub < 2; ++sub) {
      const int kbs = kb + sub * 16;
      // B frag of K^T: lane n = key, feat = kc + 16*hl + e (contiguous)
      const half_t* kp = qkv + (size_t)(b * 1024 + kbs + m16) * 768 + 256 + h * 64 + hl * 16;
      const v16h bk0 = cat8(*(const v8h*)(kp), *(const v8h*)(kp + 8));
      const v16h bk1 = cat8(*(const v8h*)(kp + 32), *(const v8h*)(kp + 40));
      v8f c = {};
      c = wmma16(aq[0], bk0, c);
      c = wmma16(aq[1], bk1, c);
      st[sub] = c;
    }
    // band mask + scale
#pragma unroll
    for (int sub = 0; sub < 2; ++sub)
#pragma unroll
      for (int r = 0; r < 8; ++r) {
        const int qi = q0 + r + hl * 8;
        const int ki = kb + sub * 16 + m16;
        float s = st[sub][r] * 0.125f;  // hd^-0.5, hd=64
        const bool ok = (qi >= ki) && (qi - ki < 256);
        pbuf[sub][r] = ok ? s : -1e30f;
      }
    // online softmax (row = 16 lanes within half-wave)
    float alpha[8];
#pragma unroll
    for (int r = 0; r < 8; ++r) {
      float mrow = fmaxf(pbuf[0][r], pbuf[1][r]);
#pragma unroll
      for (int o = 8; o; o >>= 1) mrow = fmaxf(mrow, __shfl_xor(mrow, o, 16));
      const float mnew = fmaxf(rmax[r], mrow);
      alpha[r] = __expf(rmax[r] - mnew);
      rmax[r] = mnew;
    }
#pragma unroll
    for (int r = 0; r < 8; ++r) {
      o0[r] *= alpha[r]; o1[r] *= alpha[r]; o2[r] *= alpha[r]; o3[r] *= alpha[r];
      rsum[r] *= alpha[r];
      float ps = 0.f;
#pragma unroll
      for (int sub = 0; sub < 2; ++sub) {
        const float s = pbuf[sub][r];
        const float p = (s < -1e29f) ? 0.f : __expf(s - rmax[r]);
        pbuf[sub][r] = p;
        ps += p;
      }
#pragma unroll
      for (int o = 8; o; o >>= 1) ps += __shfl_xor(ps, o, 16);
      rsum[r] += ps;
    }
    // C-layout probabilities -> A-layout via LDS
    __syncthreads();
#pragma unroll
    for (int sub = 0; sub < 2; ++sub)
#pragma unroll
      for (int r = 0; r < 8; ++r)
        Pl[r + hl * 8][sub * 16 + m16] = (half_t)pbuf[sub][r];
    __syncthreads();
    const v16h pa = cat8(*(const v8h*)&Pl[m16][hl * 16],
                         *(const v8h*)&Pl[m16][hl * 16 + 8]);
    // PV: B frag from Vt (feat-major): lane n = feat, key = kb + 16*hl + e
#pragma unroll
    for (int j = 0; j < 4; ++j) {
      const half_t* vp = Vt + ((size_t)(b * 256 + h * 64 + j * 16 + m16)) * 1024 + kb + hl * 16;
      const v16h bv = cat8(*(const v8h*)vp, *(const v8h*)(vp + 8));
      v8f& oo = (j == 0 ? o0 : j == 1 ? o1 : j == 2 ? o2 : o3);
      oo = wmma16(pa, bv, oo);
    }
    __syncthreads();
  }
  // write pre-projection attention (f16) for the Wo GEMM
#pragma unroll
  for (int r = 0; r < 8; ++r) {
    const float inv = 1.0f / rsum[r];
    const size_t row = (size_t)(b * 1024 + q0 + r + hl * 8);
    attn16[row * 256 + h * 64 + 0 * 16 + m16] = (half_t)(o0[r] * inv);
    attn16[row * 256 + h * 64 + 1 * 16 + m16] = (half_t)(o1[r] * inv);
    attn16[row * 256 + h * 64 + 2 * 16 + m16] = (half_t)(o2[r] * inv);
    attn16[row * 256 + h * 64 + 3 * 16 + m16] = (half_t)(o3[r] * inv);
  }
}

// --------------------------------------------- Atlas sequential memory scan --
// Column-separable: state W/Wp/Sm[:,v] kept in VGPRs. 256 threads = 8 waves,
// wave owns 4 columns, lane owns rows {lane+32*i}. grid = (DV/32, B).
__global__ __launch_bounds__(256) void atlas_scan_kernel(
    const float* __restrict__ km, const float* __restrict__ vm,
    const float* __restrict__ qm,
    const float* __restrict__ llog, const float* __restrict__ glog,
    const float* __restrict__ rlog,
    float* __restrict__ memout, half_t* __restrict__ memout16, int ld16) {
  __shared__ float kq[2][2][256];
  const int b = blockIdx.y;
  const int c0 = blockIdx.x * 32;
  const int tid = threadIdx.x;
  const int lane = tid & 31, wid = tid >> 5;
  const int cbase = c0 + wid * 4;
  const float lamL = 1.f / (1.f + __expf(-llog[0]));
  const float lamG = 1.f / (1.f + __expf(-glog[0]));
  const float lr = 1.f / (1.f + __expf(-rlog[0]));

  float W[4][8], Wp[4][8], Sm[4][8];
#pragma unroll
  for (int c = 0; c < 4; ++c)
#pragma unroll
    for (int i = 0; i < 8; ++i) { W[c][i] = 0.f; Wp[c][i] = 0.f; Sm[c][i] = 0.f; }

  const size_t rowbase = (size_t)b * 1024 * 256;
  kq[0][0][tid] = km[rowbase + tid];
  kq[0][1][tid] = qm[rowbase + tid];
  __syncthreads();

  for (int t = 0; t < 1024; ++t) {
    const int buf = t & 1;
    if (t + 1 < 1024) {  // double-buffer next k_t/q_t
      kq[buf ^ 1][0][tid] = km[rowbase + (size_t)(t + 1) * 256 + tid];
      kq[buf ^ 1][1][tid] = qm[rowbase + (size_t)(t + 1) * 256 + tid];
    }
    float kv[8], qv[8];
#pragma unroll
    for (int i = 0; i < 8; ++i) {
      kv[i] = kq[buf][0][lane + 32 * i];
      qv[i] = kq[buf][1][lane + 32 * i];
    }
#pragma unroll
    for (int c = 0; c < 4; ++c) {
      float pred = 0.f;
#pragma unroll
      for (int i = 0; i < 8; ++i) pred += kv[i] * W[c][i];
#pragma unroll
      for (int o = 16; o; o >>= 1) pred += __shfl_xor(pred, o, 32);
      const float err = pred - vm[rowbase + (size_t)t * 256 + cbase + c];
      float outp = 0.f;
#pragma unroll
      for (int i = 0; i < 8; ++i) {
        const float g = kv[i] * err;                       // grad
        const float sn = 0.9f * Sm[c][i] + g;              // momentum (BETA)
        const float w = W[c][i];
        const float wn = w - lr * (sn + 2.f * lamL * (w - Wp[c][i]) + 2.f * lamG * w);
        Wp[c][i] = w; W[c][i] = wn; Sm[c][i] = sn;
        outp += qv[i] * wn;                                // read after write
      }
#pragma unroll
      for (int o = 16; o; o >>= 1) outp += __shfl_xor(outp, o, 32);
      if (lane == 0) {
        memout[rowbase + (size_t)t * 256 + cbase + c] = outp;
        memout16[((size_t)b * 1024 + t) * ld16 + cbase + c] = (half_t)outp;
      }
    }
    __syncthreads();
  }
}

// ------------------------------------------------------------- elementwise --
__global__ __launch_bounds__(256) void gate_combine_kernel(
    const float* __restrict__ x, const float* __restrict__ gl,
    const float* __restrict__ mem, const float* __restrict__ attn,
    float* __restrict__ x1) {
  const int i = blockIdx.x * 256 + threadIdx.x;
  const float g = 1.f / (1.f + __expf(-gl[i]));
  x1[i] = x[i] + g * mem[i] + (1.f - g) * attn[i];
}

__global__ __launch_bounds__(256) void gelu_kernel(const float* __restrict__ in,
                                                   half_t* __restrict__ out) {
  const int i = blockIdx.x * 256 + threadIdx.x;
  const float v = in[i];
  const float c = 0.7978845608f * (v + 0.044715f * v * v * v);
  out[i] = (half_t)(0.5f * v * (1.f + tanhf(c)));
}

__global__ __launch_bounds__(256) void add_kernel(const float* __restrict__ a,
                                                  const float* __restrict__ b,
                                                  float* __restrict__ o) {
  const int i = blockIdx.x * 256 + threadIdx.x;
  o[i] = a[i] + b[i];
}

// ---------------------------------------------------------------- launcher --
extern "C" void kernel_launch(void* const* d_in, const int* in_sizes, int n_in,
                              void* d_out, int out_size, void* d_ws, size_t ws_size,
                              hipStream_t stream) {
  const float* x      = (const float*)d_in[0];
  const float* ln1_g  = (const float*)d_in[1];
  const float* ln1_b  = (const float*)d_in[2];
  const float* Wqkv   = (const float*)d_in[3];
  const float* bqkv   = (const float*)d_in[4];
  const float* Wo     = (const float*)d_in[5];
  const float* bo     = (const float*)d_in[6];
  const float* ln2_g  = (const float*)d_in[7];
  const float* ln2_b  = (const float*)d_in[8];
  const float* Wk     = (const float*)d_in[9];
  const float* Wv     = (const float*)d_in[10];
  const float* Wq     = (const float*)d_in[11];
  const float* llog   = (const float*)d_in[12];
  const float* glog   = (const float*)d_in[13];
  const float* rlog   = (const float*)d_in[14];
  const float* gate_W = (const float*)d_in[15];
  const float* gate_b = (const float*)d_in[16];
  const float* ln3_g  = (const float*)d_in[17];
  const float* ln3_b  = (const float*)d_in[18];
  const float* fc1_W  = (const float*)d_in[19];
  const float* fc1_b  = (const float*)d_in[20];
  const float* fc2_W  = (const float*)d_in[21];
  const float* fc2_b  = (const float*)d_in[22];
  float* out = (float*)d_out;

  constexpr int M = 2048, D = 256, DFF = 1024;

  uint8_t* ws = (uint8_t*)d_ws;
  size_t off = 0;
  auto alloc = [&](size_t bytes) -> void* {
    void* p = ws + off;
    off += (bytes + 255) & ~(size_t)255;
    return p;
  };
  half_t* h1       = (half_t*)alloc((size_t)M * D * 2);
  half_t* h2       = (half_t*)alloc((size_t)M * D * 2);
  half_t* qkv16    = (half_t*)alloc((size_t)M * 3 * D * 2);
  half_t* Vt       = (half_t*)alloc((size_t)2 * 256 * 1024 * 2);
  half_t* attnpre  = (half_t*)alloc((size_t)M * D * 2);
  half_t* gatein   = (half_t*)alloc((size_t)M * 512 * 2);  // [mem16 | attn16]
  float*  kmb      = (float*)alloc((size_t)M * D * 4);
  float*  vmb      = (float*)alloc((size_t)M * D * 4);
  float*  qmb      = (float*)alloc((size_t)M * D * 4);
  float*  memout   = (float*)alloc((size_t)M * D * 4);
  float*  attnproj = (float*)alloc((size_t)M * D * 4);
  float*  gatelin  = (float*)alloc((size_t)M * D * 4);
  float*  x1       = (float*)alloc((size_t)M * D * 4);
  half_t* h3       = (half_t*)alloc((size_t)M * D * 2);
  float*  ff1      = (float*)alloc((size_t)M * DFF * 4);
  half_t* h4       = (half_t*)alloc((size_t)M * DFF * 2);
  float*  ff2      = (float*)alloc((size_t)M * D * 4);

  const dim3 blk(256);

  // layernorms (attention + memory branches)
  ln_f16_kernel<<<M, blk, 0, stream>>>(x, ln1_g, ln1_b, h1);
  ln_f16_kernel<<<M, blk, 0, stream>>>(x, ln2_g, ln2_b, h2);

  // qkv projection -> f16
  gemm_wmma_kernel<<<dim3(768 / 64, M / 64), blk, 0, stream>>>(
      h1, 256, Wqkv, 768, bqkv, nullptr, 0, qkv16, 768, M, 768, 256);
  vtrans_kernel<<<(2 * 256 * 1024) / 256, blk, 0, stream>>>(qkv16, Vt);

  // sliding-window attention (one wave / 16 queries)
  attn_kernel<<<dim3(64, 4, 2), dim3(32), 0, stream>>>(qkv16, Vt, attnpre);

  // output projection: f32 for combine, f16 into right half of gate input
  gemm_wmma_kernel<<<dim3(4, 32), blk, 0, stream>>>(
      attnpre, 256, Wo, 256, bo, attnproj, 256, gatein + 256, 512, M, 256, 256);

  // memory projections (f32 for the scan)
  gemm_wmma_kernel<<<dim3(4, 32), blk, 0, stream>>>(
      h2, 256, Wk, 256, nullptr, kmb, 256, nullptr, 0, M, 256, 256);
  gemm_wmma_kernel<<<dim3(4, 32), blk, 0, stream>>>(
      h2, 256, Wv, 256, nullptr, vmb, 256, nullptr, 0, M, 256, 256);
  gemm_wmma_kernel<<<dim3(4, 32), blk, 0, stream>>>(
      h2, 256, Wq, 256, nullptr, qmb, 256, nullptr, 0, M, 256, 256);

  // sequential Atlas scan (column-parallel), f16 copy into left half of gatein
  atlas_scan_kernel<<<dim3(8, 2), blk, 0, stream>>>(
      kmb, vmb, qmb, llog, glog, rlog, memout, gatein, 512);

  // gate projection on concat([mem, attn])
  gemm_wmma_kernel<<<dim3(4, 32), blk, 0, stream>>>(
      gatein, 512, gate_W, 256, gate_b, gatelin, 256, nullptr, 0, M, 256, 512);
  gate_combine_kernel<<<M, blk, 0, stream>>>(x, gatelin, memout, attnproj, x1);

  // FFN
  ln_f16_kernel<<<M, blk, 0, stream>>>(x1, ln3_g, ln3_b, h3);
  gemm_wmma_kernel<<<dim3(16, 32), blk, 0, stream>>>(
      h3, 256, fc1_W, 1024, fc1_b, ff1, 1024, nullptr, 0, M, 1024, 256);
  gelu_kernel<<<(M * DFF) / 256, blk, 0, stream>>>(ff1, h4);
  gemm_wmma_kernel<<<dim3(4, 32), blk, 0, stream>>>(
      h4, 1024, fc2_W, 256, fc2_b, ff2, 256, nullptr, 0, M, 256, 1024);

  // final residual
  add_kernel<<<M, blk, 0, stream>>>(x1, ff2, out);
}
